// TargetEdgeInitializer_4191888081048
// MI455X (gfx1250) — matrix operand order repaired and checked by reading, not compiled, not run on hardware
//
#include <hip/hip_runtime.h>

#define N_NODES 20000
#define N_EDGES 320000
#define D_IN    256
#define D_OUT   256
#define NQKVS   1024   // q|k|v|skip concatenated along output dim

typedef __attribute__((ext_vector_type(16))) __bf16 v16bf;
typedef __attribute__((ext_vector_type(8)))  __bf16 v8bf;
typedef __attribute__((ext_vector_type(8)))  float  v8f;

// ---------- helpers ----------
static __device__ __forceinline__ unsigned short f2bf(float f) {
  unsigned u = __float_as_uint(f);
  u += 0x7FFFu + ((u >> 16) & 1u);        // round-to-nearest-even
  return (unsigned short)(u >> 16);
}
// monotonic float<->uint mapping for atomic min/max on arbitrary-sign floats
static __device__ __forceinline__ unsigned ordf(float f) {
  unsigned u = __float_as_uint(f);
  return (u >> 31) ? ~u : (u | 0x80000000u);
}
static __device__ __forceinline__ float unordf(unsigned u) {
  unsigned b = (u >> 31) ? (u & 0x7FFFFFFFu) : ~u;
  return __uint_as_float(b);
}

// ---------- weight prep: transpose + bf16-convert Wq|Wk|Wv|Wskip into Wt[1024][256] ----------
__global__ void k_prep_w(const float* Wq, const float* Wk, const float* Wv, const float* Ws,
                         const float* bq, const float* bk, const float* bv, const float* bs,
                         unsigned short* Wt, float* bias) {
  int t = blockIdx.x * blockDim.x + threadIdx.x;      // 1024*256 threads
  int o = t >> 8, i = t & 255;
  const float* W = (o < 256) ? Wq : (o < 512) ? Wk : (o < 768) ? Wv : Ws;
  int oc = o & 255;
  Wt[(size_t)o * D_IN + i] = f2bf(W[(size_t)i * 256 + oc]);
  if (i == 0) {
    const float* B = (o < 256) ? bq : (o < 512) ? bk : (o < 768) ? bv : bs;
    bias[o] = B[oc];
  }
}

__global__ void k_convx(const float* x, unsigned short* xb, int n) {
  int t = blockIdx.x * blockDim.x + threadIdx.x;
  if (t < n) xb[t] = f2bf(x[t]);
}

// ---------- fused QKVS GEMM: [20000x256] x [256x1024] via v_wmma_f32_16x16x32_bf16 ----------
// 4 waves/block, each wave: one 16-row M tile x 64-col N strip (A-fragment reuse x4)
__global__ void k_gemm_qkvs(const unsigned short* __restrict__ xb,
                            const unsigned short* __restrict__ Wt,
                            const float* __restrict__ bias,
                            float* __restrict__ out_all) {
  int lane = threadIdx.x & 31;
  int wid  = threadIdx.x >> 5;
  int mtile = blockIdx.y * 4 + wid;
  if (mtile >= N_NODES / 16) return;                  // wave-uniform exit
  int mbase = mtile * 16;
  int nb0   = blockIdx.x * 64;
  int m    = lane & 15;
  int half = lane >> 4;

  v8f acc0 = {}, acc1 = {}, acc2 = {}, acc3 = {};
  const unsigned short* arow = xb + (size_t)(mbase + m) * D_IN;
#pragma unroll
  for (int kt = 0; kt < D_IN / 32; ++kt) {
    int kb = kt * 32;
    v16bf a = *(const v16bf*)(arow + kb + half * 16);
    int bk0 = kb + half * 16;
    v16bf b0 = *(const v16bf*)(Wt + (size_t)(nb0 +      m) * D_IN + bk0);
    v16bf b1 = *(const v16bf*)(Wt + (size_t)(nb0 + 16 + m) * D_IN + bk0);
    v16bf b2 = *(const v16bf*)(Wt + (size_t)(nb0 + 32 + m) * D_IN + bk0);
    v16bf b3 = *(const v16bf*)(Wt + (size_t)(nb0 + 48 + m) * D_IN + bk0);
    acc0 = __builtin_amdgcn_wmma_f32_16x16x32_bf16(false, a, false, b0, (short)0, acc0, false, false);
    acc1 = __builtin_amdgcn_wmma_f32_16x16x32_bf16(false, a, false, b1, (short)0, acc1, false, false);
    acc2 = __builtin_amdgcn_wmma_f32_16x16x32_bf16(false, a, false, b2, (short)0, acc2, false, false);
    acc3 = __builtin_amdgcn_wmma_f32_16x16x32_bf16(false, a, false, b3, (short)0, acc3, false, false);
  }
  int n0 = nb0 + m;
  float bb0 = bias[n0], bb1 = bias[n0 + 16], bb2 = bias[n0 + 32], bb3 = bias[n0 + 48];
#pragma unroll
  for (int r = 0; r < 8; ++r) {
    float* o = out_all + (size_t)(mbase + half * 8 + r) * NQKVS;
    o[n0]      = acc0[r] + bb0;
    o[n0 + 16] = acc1[r] + bb1;
    o[n0 + 32] = acc2[r] + bb2;
    o[n0 + 48] = acc3[r] + bb3;
  }
}

// ---------- out := skip part of out_all ----------
__global__ void k_initout(const float* out_all, float* out, int n) {
  int t = blockIdx.x * blockDim.x + threadIdx.x;
  if (t < n) {
    int node = t >> 8, c = t & 255;
    out[t] = out_all[(size_t)node * NQKVS + 768 + c];
  }
}

// ---------- attention logits: wave per edge, lane = 8 channels, head = lane/8 ----------
__global__ void k_alpha(const long long* __restrict__ ei, const float* __restrict__ edge_attr,
                        const float* __restrict__ We, const float* __restrict__ out_all,
                        float* __restrict__ alpha, unsigned* __restrict__ amax) {
  int wid = threadIdx.x >> 5, lane = threadIdx.x & 31;
  int e = blockIdx.x * 8 + wid;                     // 40000 blocks * 8 waves == 320000
  int src = (int)ei[e];
  int dst = (int)ei[N_EDGES + e];
  int c0 = lane * 8;
  float ea = edge_attr[e];
  const float* q  = out_all + (size_t)dst * NQKVS + c0;
  const float* kk = out_all + (size_t)src * NQKVS + 256 + c0;
  const float* we = We + c0;
  float p = 0.f;
#pragma unroll
  for (int j = 0; j < 8; ++j) p += q[j] * (kk[j] + ea * we[j]);
  p += __shfl_xor(p, 1, 32);
  p += __shfl_xor(p, 2, 32);
  p += __shfl_xor(p, 4, 32);
  float a = p * 0.125f;                              // 1/sqrt(64)
  if ((lane & 7) == 0) {
    int h = lane >> 3;
    alpha[e * 4 + h] = a;
    atomicMax(&amax[dst * 4 + h], ordf(a));
  }
}

// ---------- exp + segment sum ----------
__global__ void k_exp(const long long* __restrict__ ei, float* __restrict__ alpha,
                      const unsigned* __restrict__ amax, float* __restrict__ asum, int n) {
  int t = blockIdx.x * blockDim.x + threadIdx.x;
  if (t >= n) return;
  int e = t >> 2, h = t & 3;
  int dst = (int)ei[N_EDGES + e];
  float m = unordf(amax[dst * 4 + h]);
  float ex = __expf(alpha[t] - m);
  alpha[t] = ex;
  atomicAdd(&asum[dst * 4 + h], ex);
}

// ---------- weighted message scatter ----------
__global__ void k_scatter(const long long* __restrict__ ei, const float* __restrict__ edge_attr,
                          const float* __restrict__ We, const float* __restrict__ out_all,
                          const float* __restrict__ alpha, const float* __restrict__ asum,
                          float* __restrict__ out) {
  int wid = threadIdx.x >> 5, lane = threadIdx.x & 31;
  int e = blockIdx.x * 8 + wid;
  int src = (int)ei[e];
  int dst = (int)ei[N_EDGES + e];
  int h = lane >> 3;
  int c0 = lane * 8;
  float att = alpha[e * 4 + h] / (asum[dst * 4 + h] + 1e-16f);
  float ea = edge_attr[e];
  const float* v  = out_all + (size_t)src * NQKVS + 512 + c0;
  const float* we = We + c0;
  float* o = out + (size_t)dst * 256 + c0;
#pragma unroll
  for (int j = 0; j < 8; ++j)
    atomicAdd(&o[j], (v[j] + ea * we[j]) * att);
}

// ---------- GraphNorm column statistics ----------
__global__ void k_colsum(const float* __restrict__ out, float* __restrict__ colsum) {
  int c = threadIdx.x;
  float s = 0.f;
  for (int r = blockIdx.x; r < N_NODES; r += gridDim.x) s += out[(size_t)r * 256 + c];
  atomicAdd(&colsum[c], s);
}
__global__ void k_mean(const float* colsum, float* mean) {
  int c = threadIdx.x;
  mean[c] = colsum[c] * (1.0f / N_NODES);
}
__global__ void k_colsq(const float* __restrict__ out, const float* __restrict__ mean,
                        const float* __restrict__ gms, float* __restrict__ colsq) {
  int c = threadIdx.x;
  float mu = gms[c] * mean[c];
  float s = 0.f;
  for (int r = blockIdx.x; r < N_NODES; r += gridDim.x) {
    float d = out[(size_t)r * 256 + c] - mu;
    s += d * d;
  }
  atomicAdd(&colsq[c], s);
}

// ---------- normalize + ReLU, write Y transposed in bf16 for the Gram WMMA ----------
__global__ void k_norm(const float* __restrict__ out, const float* __restrict__ mean,
                       const float* __restrict__ colsq, const float* __restrict__ gw,
                       const float* __restrict__ gb, const float* __restrict__ gms,
                       unsigned short* __restrict__ ybT, int n) {
  int t = blockIdx.x * blockDim.x + threadIdx.x;
  if (t >= n) return;
  int node = t >> 8, c = t & 255;
  float mu  = gms[c] * mean[c];
  float inv = rsqrtf(colsq[c] * (1.0f / N_NODES) + 1e-5f);
  float y = gw[c] * (out[t] - mu) * inv + gb[c];
  y = fmaxf(y, 0.f);
  ybT[(size_t)c * N_NODES + node] = f2bf(y);
}

// ---------- Gram: Y^T Y, K split into 25 chunks of 800 rows, bf16 WMMA + f32 atomics ----------
__global__ void k_gram(const unsigned short* __restrict__ ybT, float* __restrict__ gram) {
  int lane = threadIdx.x;
  int mb = blockIdx.x * 16, nb = blockIdx.y * 16;
  int kstart = blockIdx.z * 800;
  int m = lane & 15, half = lane >> 4;
  const unsigned short* ar = ybT + (size_t)(mb + m) * N_NODES;   // A: row of Y^T
  const unsigned short* br = ybT + (size_t)(nb + m) * N_NODES;   // B: column n of Y
  v8f acc = {};
  for (int it = 0; it < 25; ++it) {
    int kb = kstart + it * 32;
    v16bf a = *(const v16bf*)(ar + kb + half * 16);
    v16bf b = *(const v16bf*)(br + kb + half * 16);
    acc = __builtin_amdgcn_wmma_f32_16x16x32_bf16(false, a, false, b, (short)0, acc, false, false);
  }
#pragma unroll
  for (int r = 0; r < 8; ++r)
    atomicAdd(&gram[(size_t)(mb + half * 8 + r) * 256 + nb + m], acc[r]);
}

// ---------- global min/max then normalize into d_out ----------
__global__ void k_minmax(const float* __restrict__ gram, unsigned* __restrict__ mm) {
  __shared__ unsigned smin[256], smax[256];
  int t = threadIdx.x;
  unsigned lmin = 0xFFFFFFFFu, lmax = 0u;
  for (int i = blockIdx.x * 256 + t; i < 256 * 256; i += gridDim.x * 256) {
    unsigned o = ordf(gram[i]);
    lmin = (o < lmin) ? o : lmin;
    lmax = (o > lmax) ? o : lmax;
  }
  smin[t] = lmin; smax[t] = lmax;
  __syncthreads();
  for (int s = 128; s > 0; s >>= 1) {
    if (t < s) {
      if (smin[t + s] < smin[t]) smin[t] = smin[t + s];
      if (smax[t + s] > smax[t]) smax[t] = smax[t + s];
    }
    __syncthreads();
  }
  if (t == 0) { atomicMin(&mm[0], smin[0]); atomicMax(&mm[1], smax[0]); }
}
__global__ void k_final(const float* __restrict__ gram, const unsigned* __restrict__ mm,
                        float* __restrict__ outp) {
  int t = blockIdx.x * blockDim.x + threadIdx.x;
  if (t < 256 * 256) {
    float mn = unordf(mm[0]), mx = unordf(mm[1]);
    outp[t] = (gram[t] - mn) / (mx - mn + 1e-8f);
  }
}

extern "C" void kernel_launch(void* const* d_in, const int* in_sizes, int n_in,
                              void* d_out, int out_size, void* d_ws, size_t ws_size,
                              hipStream_t stream) {
  const float*     x   = (const float*)d_in[0];
  const long long* ei  = (const long long*)d_in[1];
  const float*     ea  = (const float*)d_in[2];
  const float*     Wq  = (const float*)d_in[3];
  const float*     bq  = (const float*)d_in[4];
  const float*     Wk  = (const float*)d_in[5];
  const float*     bk  = (const float*)d_in[6];
  const float*     Wv  = (const float*)d_in[7];
  const float*     bv  = (const float*)d_in[8];
  const float*     We  = (const float*)d_in[9];
  const float*     Ws  = (const float*)d_in[10];
  const float*     bs  = (const float*)d_in[11];
  const float*     gw  = (const float*)d_in[12];
  const float*     gb  = (const float*)d_in[13];
  const float*     gms = (const float*)d_in[14];
  float* outp = (float*)d_out;

  // workspace carve-out (~130 MB), 256B aligned
  char* p = (char*)d_ws;
  auto carve = [&](size_t bytes) { char* r = p; p += (bytes + 255) & ~(size_t)255; return r; };
  unsigned short* xb      = (unsigned short*)carve((size_t)N_NODES * D_IN * 2);
  unsigned short* Wt      = (unsigned short*)carve((size_t)NQKVS * D_IN * 2);
  float*          bias    = (float*)carve((size_t)NQKVS * 4);
  float*          out_all = (float*)carve((size_t)N_NODES * NQKVS * 4);
  float*          alpha   = (float*)carve((size_t)N_EDGES * 4 * 4);
  unsigned*       amax    = (unsigned*)carve((size_t)N_NODES * 4 * 4);
  float*          asum    = (float*)carve((size_t)N_NODES * 4 * 4);
  float*          out     = (float*)carve((size_t)N_NODES * 256 * 4);
  float*          colsum  = (float*)carve(256 * 4);
  float*          colsq   = (float*)carve(256 * 4);
  float*          mean    = (float*)carve(256 * 4);
  unsigned short* ybT     = (unsigned short*)carve((size_t)256 * N_NODES * 2);
  float*          gram    = (float*)carve((size_t)256 * 256 * 4);
  unsigned*       mm      = (unsigned*)carve(2 * 4);

  // zero/init scratch (graph-capturable async memsets)
  hipMemsetAsync(amax,   0,    (size_t)N_NODES * 4 * 4, stream);
  hipMemsetAsync(asum,   0,    (size_t)N_NODES * 4 * 4, stream);
  hipMemsetAsync(colsum, 0,    256 * 4, stream);
  hipMemsetAsync(colsq,  0,    256 * 4, stream);
  hipMemsetAsync(gram,   0,    (size_t)256 * 256 * 4, stream);
  hipMemsetAsync(mm,     0xFF, 4, stream);   // ordered-min sentinel
  hipMemsetAsync(mm + 1, 0,    4, stream);   // ordered-max sentinel

  k_prep_w<<<1024, 256, 0, stream>>>(Wq, Wk, Wv, Ws, bq, bk, bv, bs, Wt, bias);
  k_convx<<<(N_NODES * D_IN) / 256, 256, 0, stream>>>(x, xb, N_NODES * D_IN);
  k_gemm_qkvs<<<dim3(16, 313), 128, 0, stream>>>(xb, Wt, bias, out_all);
  k_initout<<<(N_NODES * 256) / 256, 256, 0, stream>>>(out_all, out, N_NODES * 256);
  k_alpha<<<N_EDGES / 8, 256, 0, stream>>>(ei, ea, We, out_all, alpha, amax);
  k_exp<<<(N_EDGES * 4) / 256, 256, 0, stream>>>(ei, alpha, amax, asum, N_EDGES * 4);
  k_scatter<<<N_EDGES / 8, 256, 0, stream>>>(ei, ea, We, out_all, alpha, asum, out);
  k_colsum<<<128, 256, 0, stream>>>(out, colsum);
  k_mean<<<1, 256, 0, stream>>>(colsum, mean);
  k_colsq<<<128, 256, 0, stream>>>(out, mean, gms, colsq);
  k_norm<<<(N_NODES * 256) / 256, 256, 0, stream>>>(out, mean, colsq, gw, gb, gms, ybT, N_NODES * 256);
  k_gram<<<dim3(16, 16, 25), 32, 0, stream>>>(ybT, gram);
  k_minmax<<<64, 256, 0, stream>>>(gram, mm);
  k_final<<<256, 256, 0, stream>>>(gram, mm, outp);
}
